// Attention_48601849922045
// MI455X (gfx1250) — matrix operand order repaired and checked by reading, not compiled, not run on hardware
//
#include <hip/hip_runtime.h>

// ---------------------------------------------------------------------------
// GQA attention layer for MI455X (gfx1250, wave32, WMMA bf16 16x16x32).
// Pipeline: cvt -> QKV GEMMs -> RoPE/transpose -> flash attention -> out GEMM.
// GEMM: 128x128 block tile, K-step 64, async global->LDS staging (ASYNCcnt).
// ---------------------------------------------------------------------------

typedef __attribute__((ext_vector_type(16))) __bf16 v16bf;
typedef __attribute__((ext_vector_type(8)))  float  v8f;
typedef unsigned short u16;
typedef int b128v __attribute__((__vector_size__(16)));  // async-LDS builtin pointee

constexpr int Bc  = 2;
constexpr int Sc  = 2048;
constexpr int Dc  = 2048;
constexpr int Hc  = 32;
constexpr int KVc = 8;
constexpr int HDc = 64;
constexpr int REPc = Hc / KVc;

#define DEVI __device__ __forceinline__

DEVI u16 f2bf(float f) {
  unsigned int u = __float_as_uint(f);
  unsigned int r = (u + 0x7FFFu + ((u >> 16) & 1u)) >> 16;
  return (u16)r;
}

union FragU { uint4 u[2]; v16bf v; };

// A fragment, 16x32 bf16, contiguous k order (k = 16*hi + s), wave32:
//   lane<16 : M=lane,    elems = K 0..15
//   lane>=16: M=lane-16, elems = K 16..31
DEVI v16bf load_a_frag(const u16* src, int row0, int k0, int ld) {
  const int lane = threadIdx.x & 31;
  const int m  = lane & 15;
  const int hi = lane >> 4;
  const u16* p = src + (size_t)(row0 + m) * ld + k0;
  FragU f;
  f.u[0] = *(const uint4*)(p + hi * 16);
  f.u[1] = *(const uint4*)(p + hi * 16 + 8);
  return f.v;
}

// B fragment, 32x16 bf16, where B[kk][n] = src[(col0+n)*ld + k0+kk]:
//   lane<16 : N=lane,    elems = K 0..15
//   lane>=16: N=lane-16, elems = K 16..31
DEVI v16bf load_b_frag(const u16* src, int col0, int k0, int ld) {
  const int lane = threadIdx.x & 31;
  const int n  = lane & 15;
  const int hi = lane >> 4;
  const u16* p = src + (size_t)(col0 + n) * ld + k0 + hi * 16;
  FragU f;
  f.u[0] = *(const uint4*)(p);
  f.u[1] = *(const uint4*)(p + 8);
  return f.v;
}

DEVI v8f wmma_bf(v16bf a, v16bf b, v8f c) {
  return __builtin_amdgcn_wmma_f32_16x16x32_bf16(
      /*neg_a=*/false, a, /*neg_b=*/false, b,
      /*c_mod=*/(short)0, c, /*reuse_a=*/false, /*reuse_b=*/false);
}

// 16B global -> LDS, async path when available (tracked by ASYNCcnt).
DEVI void async_copy16(const u16* g, u16* l) {
#if __has_builtin(__builtin_amdgcn_global_load_async_to_lds_b128)
  __builtin_amdgcn_global_load_async_to_lds_b128(
      (__attribute__((address_space(1))) b128v*)g,
      (__attribute__((address_space(3))) b128v*)l, 0, 0);
#else
  *(uint4*)l = *(const uint4*)g;
#endif
}
DEVI void wait_async() {
#if __has_builtin(__builtin_amdgcn_global_load_async_to_lds_b128)
  asm volatile("s_wait_asynccnt 0x0" ::: "memory");
#endif
}

DEVI float half16_max(float v) {
#pragma unroll
  for (int m = 1; m < 16; m <<= 1) v = fmaxf(v, __shfl_xor(v, m, 32));
  return v;
}
DEVI float half16_sum(float v) {
#pragma unroll
  for (int m = 1; m < 16; m <<= 1) v += __shfl_xor(v, m, 32);
  return v;
}

// ---------------------------------------------------------------------------
__global__ void cvt_f32_bf16(const float* __restrict__ in, u16* __restrict__ out, int n) {
  int i = blockIdx.x * 256 + threadIdx.x;
  if (i < n) out[i] = f2bf(in[i]);
}

// RoPE + downcast: in f32 [B,S,nh,HD] -> out bf16 same layout
__global__ void rope_cvt(const float* __restrict__ in, const float* __restrict__ fc,
                         u16* __restrict__ out, int nh) {
  const int half = HDc / 2;
  int idx = blockIdx.x * 256 + threadIdx.x;
  int total = Bc * Sc * nh * half;
  if (idx >= total) return;
  int j = idx % half;
  int h = (idx / half) % nh;
  int s = (idx / (half * nh)) % Sc;
  int b = idx / (half * nh * Sc);
  float c  = fc[(s * half + j) * 2 + 0];
  float sn = fc[(s * half + j) * 2 + 1];
  size_t base = (((size_t)(b * Sc + s) * nh + h) * HDc) + 2 * j;
  float tr = in[base], ti = in[base + 1];
  out[base]     = f2bf(tr * c - ti * sn);
  out[base + 1] = f2bf(tr * sn + ti * c);
}

// V: f32 [B,S,KV,HD] -> bf16 [B,KV,HD,S]  (so PV B-fragments are contiguous)
__global__ void v_transpose(const float* __restrict__ in, u16* __restrict__ out) {
  int idx = blockIdx.x * 256 + threadIdx.x;
  int total = Bc * KVc * HDc * Sc;
  if (idx >= total) return;
  int s = idx % Sc;
  int d = (idx / Sc) % HDc;
  int g = (idx / (Sc * HDc)) % KVc;
  int b = idx / (Sc * HDc * KVc);
  float v = in[((size_t)(b * Sc + s) * KVc + g) * HDc + d];
  out[idx] = f2bf(v);
}

// ---------------------------------------------------------------------------
// C[M,N] f32 = A[M,K] bf16 * W[N,K]^T bf16.
// Block: 256 thr = 8 waves. Tile 128(M) x 128(N), K staged 64 wide via
// async global->LDS (stride 72 elems = 144B to dodge bank conflicts).
// Wave tile 32x64 -> 16 WMMAs per staging round.
__global__ __launch_bounds__(256) void gemm_nt_bf16(
    const u16* __restrict__ A, const u16* __restrict__ W, float* __restrict__ C,
    int M, int N, int K) {
  __shared__ u16 Al[128 * 72];
  __shared__ u16 Wl[128 * 72];
  const int tid  = threadIdx.x;
  const int wave = tid >> 5;
  const int wm = wave & 3, wn = wave >> 2;       // 4 x 2 waves
  const size_t mBase = (size_t)blockIdx.y * 128;
  const size_t nBase = (size_t)blockIdx.x * 128;

  v8f acc[2][4] = {};

  for (int k0 = 0; k0 < K; k0 += 64) {
    __syncthreads();
    // Stage A and W tiles: 128 rows x 64 elems each = 1024 16B-slots each;
    // 256 threads x 4 slots per tile.
#pragma unroll
    for (int j = 0; j < 4; ++j) {
      int s = tid + 256 * j;
      int row = s >> 3, chunk = s & 7;
      async_copy16(A + (mBase + row) * (size_t)K + k0 + chunk * 8,
                   Al + row * 72 + chunk * 8);
      async_copy16(W + (nBase + row) * (size_t)K + k0 + chunk * 8,
                   Wl + row * 72 + chunk * 8);
    }
    if (k0 + 64 < K)  // hint next K tile toward GL2
      __builtin_prefetch(A + (mBase + (tid >> 1)) * (size_t)K + k0 + 64, 0, 1);
    wait_async();
    __syncthreads();

#pragma unroll
    for (int kc = 0; kc < 64; kc += 32) {
      v16bf a0 = load_a_frag(Al, wm * 32,      kc, 72);
      v16bf a1 = load_a_frag(Al, wm * 32 + 16, kc, 72);
#pragma unroll
      for (int ni = 0; ni < 4; ++ni) {
        v16bf bf = load_b_frag(Wl, wn * 64 + ni * 16, kc, 72);
        acc[0][ni] = wmma_bf(a0, bf, acc[0][ni]);
        acc[1][ni] = wmma_bf(a1, bf, acc[1][ni]);
      }
    }
  }

  const int lane = tid & 31, n = lane & 15, hi = lane >> 4;
  size_t m0 = mBase + wm * 32, n0 = nBase + wn * 64;
#pragma unroll
  for (int mi = 0; mi < 2; ++mi)
#pragma unroll
    for (int ni = 0; ni < 4; ++ni)
#pragma unroll
      for (int i = 0; i < 8; ++i)
        C[(m0 + mi * 16 + i + 8 * hi) * N + n0 + ni * 16 + n] = acc[mi][ni][i];
}

// ---------------------------------------------------------------------------
// Flash attention: one wave per 16-query tile, 32-key blocks, online softmax.
// Q bf16 [B,S,H,HD], K bf16 [B,S,KV,HD], V bf16 [B,KV,HD,S] -> O f32 [B,S,H*HD]
__global__ __launch_bounds__(256) void flash_attn(
    const u16* __restrict__ Q, const u16* __restrict__ Kb,
    const u16* __restrict__ Vt, float* __restrict__ O) {
  __shared__ u16 pbuf[8 * 16 * 40];
  const int tid = threadIdx.x;
  const int wave = tid >> 5;
  const int lane = tid & 31;
  const int n = lane & 15, hi = lane >> 4;
  const int QT = Sc / 16;

  int wg = blockIdx.x * 8 + wave;
  int qtile = wg % QT;
  int bh = wg / QT;
  int b = bh / Hc, h = bh % Hc, g = h / REPc;
  int s0 = qtile * 16;

  const u16* qp = Q  + (size_t)b * Sc * (Hc * HDc)  + (size_t)h * HDc;
  const u16* kp = Kb + (size_t)b * Sc * (KVc * HDc) + (size_t)g * HDc;
  const u16* vp = Vt + ((size_t)b * KVc + g) * HDc * Sc;

  v16bf qa0 = load_a_frag(qp, s0, 0,  Hc * HDc);
  v16bf qa1 = load_a_frag(qp, s0, 32, Hc * HDc);

  v8f o0 = {}, o1 = {}, o2 = {}, o3 = {};
  float mrun[8], lrun[8];
#pragma unroll
  for (int i = 0; i < 8; ++i) { mrun[i] = -3.0e38f; lrun[i] = 0.f; }
  const float scale = 0.125f;  // 1/sqrt(64)
  u16* pw = pbuf + wave * (16 * 40);

  const int nblk = (s0 + 16 + 31) / 32;
  for (int jb = 0; jb < nblk; ++jb) {
    const int key0 = jb * 32;
    v8f z = {};
    // scores: Q(16x64) x K^T(64x32keys) -> two 16x16 tiles
    v16bf kb00 = load_b_frag(kp, key0,      0,  KVc * HDc);
    v16bf kb01 = load_b_frag(kp, key0,      32, KVc * HDc);
    v8f sv0 = wmma_bf(qa0, kb00, z);
    sv0 = wmma_bf(qa1, kb01, sv0);
    v16bf kb10 = load_b_frag(kp, key0 + 16, 0,  KVc * HDc);
    v16bf kb11 = load_b_frag(kp, key0 + 16, 32, KVc * HDc);
    v8f sv1 = wmma_bf(qa0, kb10, z);
    sv1 = wmma_bf(qa1, kb11, sv1);

#pragma unroll
    for (int i = 0; i < 8; ++i) {
      int row = s0 + i + 8 * hi;
      float a0s = sv0[i] * scale;
      float a1s = sv1[i] * scale;
      if (key0 + n > row)      a0s = -3.0e38f;   // causal mask
      if (key0 + 16 + n > row) a1s = -3.0e38f;
      float tmax = half16_max(fmaxf(a0s, a1s));
      float mnew = fmaxf(mrun[i], tmax);
      float alpha = __expf(mrun[i] - mnew);
      float e0 = __expf(a0s - mnew);
      float e1 = __expf(a1s - mnew);
      float rsum = half16_sum(e0 + e1);
      lrun[i] = lrun[i] * alpha + rsum;
      mrun[i] = mnew;
      o0[i] *= alpha; o1[i] *= alpha; o2[i] *= alpha; o3[i] *= alpha;
      int m = i + 8 * hi;                        // C-layout -> LDS (row-major P)
      pw[m * 40 + n]      = f2bf(e0);
      pw[m * 40 + 16 + n] = f2bf(e1);
    }
    asm volatile("s_wait_dscnt 0x0" ::: "memory");  // cross-lane LDS exchange
    v16bf pa = load_a_frag(pw, 0, 0, 40);           // P in A-fragment layout
    // O(16x64) += P(16x32keys) x V(32keys x 64d)
    v16bf vb;
    vb = load_b_frag(vp, 0,  key0, Sc); o0 = wmma_bf(pa, vb, o0);
    vb = load_b_frag(vp, 16, key0, Sc); o1 = wmma_bf(pa, vb, o1);
    vb = load_b_frag(vp, 32, key0, Sc); o2 = wmma_bf(pa, vb, o2);
    vb = load_b_frag(vp, 48, key0, Sc); o3 = wmma_bf(pa, vb, o3);
  }

  float* op = O + (size_t)b * Sc * (Hc * HDc) + (size_t)h * HDc;
#pragma unroll
  for (int i = 0; i < 8; ++i) {
    size_t row = (size_t)s0 + i + 8 * hi;
    float inv = 1.0f / lrun[i];
    op[row * (Hc * HDc) + n]      = o0[i] * inv;
    op[row * (Hc * HDc) + 16 + n] = o1[i] * inv;
    op[row * (Hc * HDc) + 32 + n] = o2[i] * inv;
    op[row * (Hc * HDc) + 48 + n] = o3[i] * inv;
  }
}

// ---------------------------------------------------------------------------
extern "C" void kernel_launch(void* const* d_in, const int* in_sizes, int n_in,
                              void* d_out, int out_size, void* d_ws, size_t ws_size,
                              hipStream_t stream) {
  const float* x  = (const float*)d_in[0];
  const float* fc = (const float*)d_in[1];
  const float* wq = (const float*)d_in[2];
  const float* wk = (const float*)d_in[3];
  const float* wv = (const float*)d_in[4];
  const float* wo = (const float*)d_in[5];

  const int M  = Bc * Sc;          // 4096
  const int NQ = Hc * HDc;         // 2048
  const int NK = KVc * HDc;        // 512

  char* ws = (char*)d_ws;
  size_t off = 0;
  auto carve = [&](size_t bytes) { void* p = ws + off; off += bytes; return p; };

  u16*   xb  = (u16*)  carve((size_t)M * Dc * 2);
  u16*   wqb = (u16*)  carve((size_t)NQ * Dc * 2);
  u16*   wkb = (u16*)  carve((size_t)NK * Dc * 2);
  u16*   wvb = (u16*)  carve((size_t)NK * Dc * 2);
  u16*   wob = (u16*)  carve((size_t)Dc * NQ * 2);
  float* qf  = (float*)carve((size_t)M * NQ * 4);
  float* kf  = (float*)carve((size_t)M * NK * 4);
  float* vf  = (float*)carve((size_t)M * NK * 4);
  u16*   qb  = (u16*)  carve((size_t)M * NQ * 2);
  u16*   kbf = (u16*)  carve((size_t)M * NK * 2);
  u16*   vtb = (u16*)  carve((size_t)M * NK * 2);
  float* of  = (float*)carve((size_t)M * NQ * 4);
  u16*   ob  = (u16*)  carve((size_t)M * NQ * 2);
  (void)ws_size; (void)in_sizes; (void)n_in; (void)out_size;

  auto blocks = [](size_t n) { return (unsigned)((n + 255) / 256); };

  cvt_f32_bf16<<<blocks((size_t)M * Dc),  256, 0, stream>>>(x,  xb,  M * Dc);
  cvt_f32_bf16<<<blocks((size_t)NQ * Dc), 256, 0, stream>>>(wq, wqb, NQ * Dc);
  cvt_f32_bf16<<<blocks((size_t)NK * Dc), 256, 0, stream>>>(wk, wkb, NK * Dc);
  cvt_f32_bf16<<<blocks((size_t)NK * Dc), 256, 0, stream>>>(wv, wvb, NK * Dc);
  cvt_f32_bf16<<<blocks((size_t)Dc * NQ), 256, 0, stream>>>(wo, wob, Dc * NQ);

  gemm_nt_bf16<<<dim3(NQ / 128, M / 128), 256, 0, stream>>>(xb, wqb, qf, M, NQ, Dc);
  gemm_nt_bf16<<<dim3(NK / 128, M / 128), 256, 0, stream>>>(xb, wkb, kf, M, NK, Dc);
  gemm_nt_bf16<<<dim3(NK / 128, M / 128), 256, 0, stream>>>(xb, wvb, vf, M, NK, Dc);

  rope_cvt<<<blocks((size_t)Bc * Sc * Hc  * (HDc / 2)), 256, 0, stream>>>(qf, fc, qb,  Hc);
  rope_cvt<<<blocks((size_t)Bc * Sc * KVc * (HDc / 2)), 256, 0, stream>>>(kf, fc, kbf, KVc);
  v_transpose<<<blocks((size_t)Bc * KVc * HDc * Sc), 256, 0, stream>>>(vf, vtb);

  flash_attn<<<(Bc * Hc * (Sc / 16)) / 8, 256, 0, stream>>>(qb, kbf, vtb, of);

  cvt_f32_bf16<<<blocks((size_t)M * NQ), 256, 0, stream>>>(of, ob, M * NQ);
  gemm_nt_bf16<<<dim3(Dc / 128, M / 128), 256, 0, stream>>>(ob, wob, (float*)d_out, M, Dc, NQ);
}
